// MultiHeadSelfAttention_44641890075021
// MI455X (gfx1250) — matrix operand order repaired and checked by reading, not compiled, not run on hardware
//
#include <hip/hip_runtime.h>

#define B_   2
#define S_   4096
#define E_   1024
#define H_   16
#define DQK_ 64
#define DV_  64
#define M_   (B_*S_)           // 8192 rows
#define HD_  (H_*DQK_)         // 1024
#define NKV_ (H_*(DQK_+DV_))   // 2048

typedef __attribute__((ext_vector_type(16))) __bf16 v16bf;
typedef __attribute__((ext_vector_type(8)))  float  v8f;

union FragBF {
  v16bf v;
  unsigned int u[8];
};

__device__ __forceinline__ unsigned short f2bf(float f) {
  unsigned int u = __float_as_uint(f);
  unsigned int r = u + 0x7FFFu + ((u >> 16) & 1u);   // round-to-nearest-even
  return (unsigned short)(r >> 16);
}
__device__ __forceinline__ float bf2f(unsigned short s) {
  return __uint_as_float(((unsigned int)s) << 16);
}

__device__ __forceinline__ v8f wmma_bf16(v16bf a, v16bf b, v8f c) {
  return __builtin_amdgcn_wmma_f32_16x16x32_bf16(false, a, false, b, (short)0, c,
                                                 false, false);
}

// ---- DPP16 16-lane reductions (no LDS, no counter waits) ------------------
template <int CTRL>
__device__ __forceinline__ float dpp_f(float x) {
  return __int_as_float(__builtin_amdgcn_update_dpp(
      0, __float_as_int(x), CTRL, 0xF, 0xF, true));
}
// Reduce across each 16-lane half of the wave (DPP row = 16 lanes).
__device__ __forceinline__ float red16_max(float x) {
  x = fmaxf(x, dpp_f<0xB1>(x));    // quad_perm [1,0,3,2]  (xor 1)
  x = fmaxf(x, dpp_f<0x4E>(x));    // quad_perm [2,3,0,1]  (xor 2)
  x = fmaxf(x, dpp_f<0x141>(x));   // row_half_mirror (pairs quads in 8-block)
  x = fmaxf(x, dpp_f<0x140>(x));   // row_mirror (pairs 8-blocks in row)
  return x;
}
__device__ __forceinline__ float red16_sum(float x) {
  x += dpp_f<0xB1>(x);
  x += dpp_f<0x4E>(x);
  x += dpp_f<0x141>(x);
  x += dpp_f<0x140>(x);
  return x;
}

// ---- CDNA5 fragment loaders (contiguous k order: K = 16*half + element) ----------------------
// A fragment: 16x32 bf16. Lane (half = lane>>4, r = lane&15) owns row r.
// VGPR v: K = 2v + 16*half (+0/1).
__device__ __forceinline__ void load_a_frag(FragBF& f, const unsigned short* base,
                                            int ld, int lane) {
  int half = lane >> 4, r = lane & 15;
  const unsigned short* p = base + (size_t)r * ld;
#pragma unroll
  for (int v = 0; v < 8; ++v) {
    int bk = v * 2 + half * 16;
    f.u[v] = *(const unsigned int*)(p + bk);
  }
}
// B fragment: 32x16 bf16 from an N-major buffer (row n holds K contiguously).
// Lane owns column (lane&15); lanes 0-15 -> K 0..15, lanes 16-31 -> K 16..31.
__device__ __forceinline__ void load_b_frag(FragBF& f, const unsigned short* base,
                                            int ld, int lane) {
  int half = lane >> 4, c = lane & 15;
  const unsigned short* p = base + (size_t)c * ld + half * 16;
#pragma unroll
  for (int v = 0; v < 8; ++v) f.u[v] = *(const unsigned int*)(p + v * 2);
}

// ---- conversion kernels ---------------------------------------------------
__global__ void convert_bf16(const float* __restrict__ src,
                             unsigned short* __restrict__ dst, size_t n) {
  size_t tid = (size_t)blockIdx.x * blockDim.x + threadIdx.x;
  if (tid < n) dst[tid] = f2bf(src[tid]);
}

// src [K][N] fp32 -> dst [N][K] bf16
__global__ void convert_transpose(const float* __restrict__ src,
                                  unsigned short* __restrict__ dst, int K, int N) {
  size_t tid = (size_t)blockIdx.x * blockDim.x + threadIdx.x;
  if (tid >= (size_t)K * N) return;
  int n = (int)(tid % N);
  int k = (int)(tid / N);
  dst[(size_t)n * K + k] = f2bf(src[tid]);
}

// ---- WMMA GEMM: C[M][N] = A[M][K](bf16) * Bt[N][K](bf16)^T + bias ---------
// 8 waves (2x4), wave tile 64x32 (4x2 wmma), block tile 128x128, K-step 32.
template <bool OUT_F32>
__global__ __launch_bounds__(256) void gemm_bf16_wmma(
    const unsigned short* __restrict__ A, const unsigned short* __restrict__ Bt,
    const float* __restrict__ bias, void* __restrict__ Cout, int M, int N, int K) {
  int lane = threadIdx.x & 31;
  int wave = threadIdx.x >> 5;
  int wm = wave >> 2, wn = wave & 3;
  int tm = blockIdx.x * 128 + wm * 64;
  int tn = blockIdx.y * 128 + wn * 32;
  int lr = lane & 15;
  v8f acc[4][2] = {};
  for (int k0 = 0; k0 < K; k0 += 32) {
    if (k0 + 32 < K) {  // prefetch next K slab (global_prefetch_b8)
      __builtin_prefetch(A + (size_t)(tm + lr) * K + k0 + 32, 0, 1);
      __builtin_prefetch(Bt + (size_t)(tn + lr) * K + k0 + 32, 0, 1);
    }
    FragBF a[4], b[2];
#pragma unroll
    for (int i = 0; i < 4; ++i)
      load_a_frag(a[i], A + (size_t)(tm + i * 16) * K + k0, K, lane);
#pragma unroll
    for (int j = 0; j < 2; ++j)
      load_b_frag(b[j], Bt + (size_t)(tn + j * 16) * K + k0, K, lane);
#pragma unroll
    for (int i = 0; i < 4; ++i)
#pragma unroll
      for (int j = 0; j < 2; ++j)
        acc[i][j] = wmma_bf16(a[i].v, b[j].v, acc[i][j]);
  }
  int half = lane >> 4;
#pragma unroll
  for (int i = 0; i < 4; ++i)
#pragma unroll
    for (int j = 0; j < 2; ++j) {
      int col = tn + j * 16 + lr;
      float bv = bias[col];
#pragma unroll
      for (int v = 0; v < 8; ++v) {
        int row = tm + i * 16 + v + 8 * half;
        float val = acc[i][j][v] + bv;
        if (OUT_F32) ((float*)Cout)[(size_t)row * N + col] = val;
        else ((unsigned short*)Cout)[(size_t)row * N + col] = f2bf(val);
      }
    }
}

// ---- RoPE + reshape -------------------------------------------------------
// qb  [B*S][H*64] bf16 -> qr [B,H,S,64] bf16 (roped, pre-scaled by 1/sqrt(dk))
// kvb [B*S][2048] bf16 -> kr [B,H,S,64] (roped), vt [B,H,64,S]
__global__ __launch_bounds__(256) void rope_reshape(
    const unsigned short* __restrict__ qb, const unsigned short* __restrict__ kvb,
    unsigned short* __restrict__ qr, unsigned short* __restrict__ kr,
    unsigned short* __restrict__ vt) {
  int tid = blockIdx.x * blockDim.x + threadIdx.x;  // B*H*S*32 threads
  int i = tid & 31;
  int s = (tid >> 5) & (S_ - 1);
  int h = (tid >> 17) & (H_ - 1);
  int b = tid >> 21;

  float freq = __powf(10000.0f, -(float)i / 32.0f);
  float ang = (float)s * freq;
  float cs = __cosf(ang), sn = __sinf(ang);

  // q (softmax scale 1/sqrt(64) folded in here)
  {
    size_t in = (size_t)(b * S_ + s) * HD_ + h * DQK_;
    float t1 = bf2f(qb[in + i]), t2 = bf2f(qb[in + i + 32]);
    size_t out = ((size_t)(b * H_ + h) * S_ + s) * DQK_;
    qr[out + i] = f2bf((t1 * cs - t2 * sn) * 0.125f);
    qr[out + i + 32] = f2bf((t1 * sn + t2 * cs) * 0.125f);
  }
  // k
  {
    size_t in = (size_t)(b * S_ + s) * NKV_ + h * DQK_;
    float t1 = bf2f(kvb[in + i]), t2 = bf2f(kvb[in + i + 32]);
    size_t out = ((size_t)(b * H_ + h) * S_ + s) * DQK_;
    kr[out + i] = f2bf(t1 * cs - t2 * sn);
    kr[out + i + 32] = f2bf(t1 * sn + t2 * cs);
  }
  // v (transpose to [B,H,64,S]; direct bf16 move)
  {
    size_t in = (size_t)(b * S_ + s) * NKV_ + HD_ + h * DV_;
    size_t base = (size_t)(b * H_ + h) * DV_ * S_;
    vt[base + (size_t)i * S_ + s] = kvb[in + i];
    vt[base + (size_t)(i + 32) * S_ + s] = kvb[in + i + 32];
  }
}

// ---- flash attention: one wave per 32 query rows --------------------------
// Per 32-key step: 4 K-frag + 4 V-frag loads feed 16 WMMAs (32 flops/byte).
// Interior key blocks are fully unmasked; only the diagonal block is masked.
// Row reductions use DPP16 (pure VALU), not LDS permutes.
__global__ __launch_bounds__(32) void flash_attn(
    const unsigned short* __restrict__ qr, const unsigned short* __restrict__ kr,
    const unsigned short* __restrict__ vt, unsigned short* __restrict__ attn) {
  __shared__ unsigned short lds[32 * 32];  // P tiles (D-layout -> A-layout)
  int lane = threadIdx.x;
  int blk = blockIdx.x;                 // B*H*(S/32)
  int qb32 = blk & (S_ / 32 - 1);
  int h = (blk >> 7) & (H_ - 1);
  int b = blk >> 11;
  int q_base = qb32 * 32;
  int half = lane >> 4, lr = lane & 15;

  const unsigned short* qbh = qr + (size_t)(b * H_ + h) * S_ * DQK_;
  const unsigned short* kbh = kr + (size_t)(b * H_ + h) * S_ * DQK_;
  const unsigned short* vbh = vt + (size_t)(b * H_ + h) * DV_ * S_;

  FragBF qf[2][2];  // [row-tile][k-step]
#pragma unroll
  for (int r = 0; r < 2; ++r)
#pragma unroll
    for (int c = 0; c < 2; ++c)
      load_a_frag(qf[r][c], qbh + (size_t)(q_base + r * 16) * DQK_ + c * 32,
                  DQK_, lane);

  float m[2][8], l[2][8];
  v8f ao[2][4] = {};
#pragma unroll
  for (int r = 0; r < 2; ++r)
#pragma unroll
    for (int v = 0; v < 8; ++v) { m[r][v] = -3e38f; l[r][v] = 0.0f; }

  // one flash step over keys [key0, key0+32); masked => apply causal mask
  auto step = [&](int key0, bool masked) {
    v8f sc[2][2] = {};  // [row-tile][key-subtile]
#pragma unroll
    for (int t = 0; t < 2; ++t) {
#pragma unroll
      for (int c = 0; c < 2; ++c) {
        FragBF kf;  // shared by both row-tiles
        load_b_frag(kf, kbh + (size_t)(key0 + t * 16) * DQK_ + c * 32, DQK_, lane);
#pragma unroll
        for (int r = 0; r < 2; ++r)
          sc[r][t] = wmma_bf16(qf[r][c].v, kf.v, sc[r][t]);
      }
    }

    float alpha[2][8];
#pragma unroll
    for (int r = 0; r < 2; ++r) {
      float rm[8];
#pragma unroll
      for (int v = 0; v < 8; ++v) {
        float s0 = sc[r][0][v];
        float s1 = sc[r][1][v];
        if (masked) {
          int qrow = q_base + r * 16 + v + 8 * half;
          s0 = ((key0 + lr) <= qrow) ? s0 : -3e38f;
          s1 = ((key0 + 16 + lr) <= qrow) ? s1 : -3e38f;
          sc[r][0][v] = s0; sc[r][1][v] = s1;
        }
        rm[v] = red16_max(fmaxf(s0, s1));  // DPP16 row reduction
      }

      float rs[8];
#pragma unroll
      for (int v = 0; v < 8; ++v) {
        float mn = fmaxf(m[r][v], rm[v]);
        alpha[r][v] = __expf(m[r][v] - mn);
        m[r][v] = mn;
        float p0 = __expf(sc[r][0][v] - mn);
        float p1 = __expf(sc[r][1][v] - mn);
        rs[v] = red16_sum(p0 + p1);        // DPP16 row reduction
        int row = r * 16 + v + 8 * half;
        lds[row * 32 + lr] = f2bf(p0);
        lds[row * 32 + 16 + lr] = f2bf(p1);
      }
#pragma unroll
      for (int v = 0; v < 8; ++v) l[r][v] = l[r][v] * alpha[r][v] + rs[v];
    }

    __syncthreads();  // single-wave WG: compiler fence, S_NOP in HW
    FragBF pf[2];
#pragma unroll
    for (int r = 0; r < 2; ++r) load_a_frag(pf[r], lds + r * 16 * 32, 32, lane);

#pragma unroll
    for (int j = 0; j < 4; ++j) {
      FragBF vf;  // shared by both row-tiles
      load_b_frag(vf, vbh + (size_t)(j * 16) * S_ + key0, S_, lane);
#pragma unroll
      for (int r = 0; r < 2; ++r) {
#pragma unroll
        for (int v = 0; v < 8; ++v) ao[r][j][v] *= alpha[r][v];
        ao[r][j] = wmma_bf16(pf[r].v, vf.v, ao[r][j]);
      }
    }
    __syncthreads();
  };

  // interior blocks: every key < every query row of this tile -> no mask
  int nkb = q_base / 32 + 1;
  for (int kb = 0; kb < nkb - 1; ++kb) step(kb * 32, false);
  // diagonal block: per-element causal mask
  step((nkb - 1) * 32, true);

  // write out [B,S,H*DV] bf16 (one reciprocal per row, not one div per elem)
  float linv[2][8];
#pragma unroll
  for (int r = 0; r < 2; ++r)
#pragma unroll
    for (int v = 0; v < 8; ++v) linv[r][v] = 1.0f / l[r][v];
#pragma unroll
  for (int r = 0; r < 2; ++r)
#pragma unroll
    for (int j = 0; j < 4; ++j) {
      int col = h * DV_ + j * 16 + lr;
#pragma unroll
      for (int v = 0; v < 8; ++v) {
        int qrow = q_base + r * 16 + v + 8 * half;
        attn[(size_t)(b * S_ + qrow) * HD_ + col] = f2bf(ao[r][j][v] * linv[r][v]);
      }
    }
}

// ---------------------------------------------------------------------------
extern "C" void kernel_launch(void* const* d_in, const int* in_sizes, int n_in,
                              void* d_out, int out_size, void* d_ws, size_t ws_size,
                              hipStream_t stream) {
  const float* x   = (const float*)d_in[0];
  const float* Wq  = (const float*)d_in[1];
  const float* bq  = (const float*)d_in[2];
  const float* Wkv = (const float*)d_in[3];
  const float* bkv = (const float*)d_in[4];
  const float* Wo  = (const float*)d_in[5];
  const float* bo  = (const float*)d_in[6];

  char* w = (char*)d_ws;
  auto take = [&](size_t elems) {
    unsigned short* p = (unsigned short*)w;
    w += ((elems * 2) + 255) & ~(size_t)255;
    return p;
  };
  unsigned short* xb   = take((size_t)M_ * E_);          // x bf16
  unsigned short* wqt  = take((size_t)HD_ * E_);         // Wq^T bf16
  unsigned short* wkvt = take((size_t)NKV_ * E_);        // Wkv^T bf16
  unsigned short* wot  = take((size_t)E_ * HD_);         // Wo^T bf16
  unsigned short* qb   = take((size_t)M_ * HD_);         // q = xWq+bq (bf16)
  unsigned short* kvb  = take((size_t)M_ * NKV_);        // kv (bf16)
  unsigned short* qr   = take((size_t)B_ * H_ * S_ * DQK_);
  unsigned short* kr   = take((size_t)B_ * H_ * S_ * DQK_);
  unsigned short* vt   = take((size_t)B_ * H_ * DV_ * S_);
  unsigned short* attn = take((size_t)M_ * HD_);

  convert_bf16<<<(M_ * E_) / 256, 256, 0, stream>>>(x, xb, (size_t)M_ * E_);
  convert_transpose<<<(E_ * HD_) / 256, 256, 0, stream>>>(Wq, wqt, E_, HD_);
  convert_transpose<<<(E_ * NKV_) / 256, 256, 0, stream>>>(Wkv, wkvt, E_, NKV_);
  convert_transpose<<<(HD_ * E_) / 256, 256, 0, stream>>>(Wo, wot, HD_, E_);

  gemm_bf16_wmma<false><<<dim3(M_ / 128, HD_ / 128), 256, 0, stream>>>(
      xb, wqt, bq, qb, M_, HD_, E_);
  gemm_bf16_wmma<false><<<dim3(M_ / 128, NKV_ / 128), 256, 0, stream>>>(
      xb, wkvt, bkv, kvb, M_, NKV_, E_);

  rope_reshape<<<(B_ * H_ * S_ * 32) / 256, 256, 0, stream>>>(qb, kvb, qr, kr, vt);

  flash_attn<<<B_ * H_ * (S_ / 32), 32, 0, stream>>>(qr, kr, vt, attn);

  gemm_bf16_wmma<true><<<dim3(M_ / 128, E_ / 128), 256, 0, stream>>>(
      attn, wot, bo, d_out, M_, E_, HD_);
}
